// MotionEnhancedAttention_81046032876133
// MI455X (gfx1250) — matrix operand order repaired and checked by reading, not compiled, not run on hardware
//
#include <hip/hip_runtime.h>

typedef __attribute__((ext_vector_type(16))) _Float16 v16h;
typedef __attribute__((ext_vector_type(8)))  _Float16 v8h;
typedef __attribute__((ext_vector_type(8)))  float    v8f;
typedef __attribute__((ext_vector_type(4)))  int      v4i;

#define B_ 8
#define C_ 256
#define N_ 4096
// softmax uses exp2((s - m) * SCALE * log2(e)),  SCALE = C^-0.5 = 1/16
#define C1f (0.0625f * 1.44269504088896340736f)

#define WMMA_F16(a, b, c) \
  __builtin_amdgcn_wmma_f32_16x16x32_f16(false, (a), false, (b), (short)0, (c), false, false)

#if __has_builtin(__builtin_amdgcn_global_load_async_to_lds_b128) && \
    __has_builtin(__builtin_amdgcn_s_wait_asynccnt)
#define HAVE_ASYNC 1
#else
#define HAVE_ASYNC 0
#endif

typedef __attribute__((address_space(1))) v4i* gv4p;  // global int4*
typedef __attribute__((address_space(3))) v4i* lv4p;  // LDS    int4*

// Build a 16-half fragment from two contiguous 8-half (b128) chunks.
__device__ inline v16h ld_frag(const _Float16* base, int o0, int o1) {
  v8h lo = *(const v8h*)(base + o0);
  v8h hi = *(const v8h*)(base + o1);
  v16h r;
#pragma unroll
  for (int u = 0; u < 8; ++u) { r[u] = lo[u]; r[8 + u] = hi[u]; }
  return r;
}

// One 16-byte global -> LDS move (async-to-LDS if the toolchain exposes it).
__device__ inline void mv_b128(const _Float16* g, _Float16* l) {
#if HAVE_ASYNC
  __builtin_amdgcn_global_load_async_to_lds_b128(
      (gv4p)(void*)g, (lv4p)(void*)l, 0, 0);
#else
  *(v8h*)l = *(const v8h*)g;
#endif
}

// ---------------------------------------------------------------------------
// Kernel 0: one-shot f32 -> f16 conversion of the three weight matrices.
// wf16 layout: [3][256][256] row-major.  grid 768 x 256 threads.
// ---------------------------------------------------------------------------
__global__ __launch_bounds__(256) void wcvt_kernel(
    const float* __restrict__ wq, const float* __restrict__ wk,
    const float* __restrict__ wv, _Float16* __restrict__ wf) {
  const int i = blockIdx.x * 256 + threadIdx.x;          // 0 .. 3*65536-1
  const float* src = (i < 65536) ? wq : (i < 131072 ? wk : wv);
  wf[i] = (_Float16)src[i & 65535];
}

// ---------------------------------------------------------------------------
// Kernel 1: fused 1x1-conv (channel-mixing GEMM) for Q, K, V in one pass.
//   q[b,d,n] = sum_c wq[d,c] x[b,c,n] + bq[d]     (k,v from motion_info)
// Output layouts (f16 workspace):
//   Q, K token-major  [b][n][c]   -> contiguous A/B fragments for S = Q^T K
//   V   channel-major [b][c][n]   -> contiguous B fragments for O = P V
// grid (N/16, B), block 128 (4 waves).  Stages x AND motion tiles once,
// then loops m = q,k,v with pre-converted f16 weights (pure b128 A-frags).
// ---------------------------------------------------------------------------
__global__ __launch_bounds__(128) void qkv_proj(
    const float* __restrict__ x, const float* __restrict__ motion,
    const _Float16* __restrict__ wf,
    const float* __restrict__ bq, const float* __restrict__ bk,
    const float* __restrict__ bv,
    _Float16* __restrict__ qws, _Float16* __restrict__ kws,
    _Float16* __restrict__ vws) {
  __shared__ _Float16 xLds[2][16 * C_];  // token-major [tok][c], 2 x 8 KB

  const int t0 = blockIdx.x * 16;
  const int b  = blockIdx.y;
  const int tid = threadIdx.x;

  // stage x and motion tiles (256 ch x 16 tok each), f32->f16, token-major
  for (int i = tid; i < 2 * 16 * C_; i += 128) {
    const int s = i >> 12;          // 0 = x, 1 = motion
    const int j = i & 4095;
    const int c = j >> 4, t = j & 15;
    const float* src = s ? motion : x;
    xLds[s][t * C_ + c] = (_Float16)src[((size_t)(b * C_ + c)) * N_ + t0 + t];
  }
  __syncthreads();

  const int wave = tid >> 5, lane = tid & 31;
  const int half = lane >> 4, l16 = lane & 15;

  for (int m = 0; m < 3; ++m) {
    const _Float16* wgt = wf + m * 65536;
    const float* bias = (m == 0) ? bq : (m == 1 ? bk : bv);
    const _Float16* xt = xLds[m == 0 ? 0 : 1];

    for (int it = 0; it < 4; ++it) {
      const int d0 = wave * 16 + it * 64;  // output-channel tile base
      v8f acc = {};
#pragma unroll
      for (int kc = 0; kc < 8; ++kc) {
        const int kb = kc * 32;
        // A fragment: f16 weight rows straight from global (two b128 loads)
        v16h a = ld_frag(wgt + (size_t)(d0 + l16) * C_ + kb,
                         half * 8, 16 + half * 8);
        // B fragment: input tile token-major in LDS (lane = column/token)
        v16h bf = ld_frag(&xt[l16 * C_ + kb], half * 8, 16 + half * 8);
        acc = WMMA_F16(a, bf, acc);
      }
      // bias add + store (D layout: VGPR r -> row d0+r+8*half, lane -> token)
#pragma unroll
      for (int r = 0; r < 8; ++r) {
        const int d   = d0 + r + 8 * half;
        const int tok = t0 + l16;
        const _Float16 o = (_Float16)(acc[r] + bias[d]);
        if (m == 0)      qws[((size_t)b * N_ + tok) * C_ + d] = o;
        else if (m == 1) kws[((size_t)b * N_ + tok) * C_ + d] = o;
        else             vws[((size_t)b * C_ + d) * N_ + tok] = o;
      }
    }
  }
}

// ---------------------------------------------------------------------------
// Kernel 2: fused flash-style attention.  grid (N/128, B), block 256 (8 waves).
// Each wave owns 16 query rows; K/V double-buffered in LDS via async-to-LDS
// and shared by all 8 waves.  Online softmax, O accumulated in 128 f32 VGPRs.
// ---------------------------------------------------------------------------
__global__ __launch_bounds__(256) void attn_kernel(
    const _Float16* __restrict__ qws, const _Float16* __restrict__ kws,
    const _Float16* __restrict__ vws, float* __restrict__ out) {
  __shared__ _Float16 kLds[2][32 * C_];   // token-major [key][c],   2 x 16 KB
  __shared__ _Float16 vLds[2][C_ * 32];   // channel-major [c][key], 2 x 16 KB
  __shared__ _Float16 pLds[8][16 * 32];   // per-wave P transpose,       8 KB

  const int b    = blockIdx.y;
  const int q0   = blockIdx.x * 128;
  const int tid  = threadIdx.x;
  const int wave = tid >> 5;
  const int lane = tid & 31;
  const int half = lane >> 4;
  const int l16  = lane & 15;

  // Preload this wave's Q A-fragments straight from global (token-major).
  v16h QA[8];
  {
    const _Float16* qr = qws + ((size_t)b * N_ + q0 + wave * 16 + l16) * C_;
#pragma unroll
    for (int kc = 0; kc < 8; ++kc)
      QA[kc] = ld_frag(qr + kc * 32, half * 8, 16 + half * 8);
  }

  // stage one 32-key tile (K token-major, V channel-major) into buffer bufi
  auto stage = [&](int kt, int bufi) {
    const _Float16* kg = kws + ((size_t)b * N_ + kt) * C_;
    _Float16* kd = kLds[bufi];
    for (int i = tid; i < (32 * C_) / 8; i += 256)
      mv_b128(kg + i * 8, kd + i * 8);
    _Float16* vd = vLds[bufi];
    for (int i = tid; i < (C_ * 32) / 8; i += 256) {
      const int c = i >> 2, p = i & 3;
      mv_b128(vws + ((size_t)b * C_ + c) * N_ + kt + p * 8, vd + c * 32 + p * 8);
    }
  };

  v8f O[16];
  float mrow[8], lrow[8];
  {
    v8f z = {};
#pragma unroll
    for (int t = 0; t < 16; ++t) O[t] = z;
#pragma unroll
    for (int r = 0; r < 8; ++r) { mrow[r] = -3.0e38f; lrow[r] = 0.f; }
  }

  stage(0, 0);  // prologue
  int buf = 0;

  for (int kt = 0; kt < N_; kt += 32) {
#if HAVE_ASYNC
    __builtin_amdgcn_s_wait_asynccnt(0);
#endif
    __syncthreads();  // current buffer resident + all waves past last compute

    if (kt + 32 < N_) stage(kt + 32, buf ^ 1);  // overlap DMA with compute

    const _Float16* kb_ = kLds[buf];
    const _Float16* vb_ = vLds[buf];

    // ---- S = Q^T K : 16 queries x 32 keys (two 16x16 WMMA chains) ----
    v8f S0 = {}, S1 = {};
#pragma unroll
    for (int kc = 0; kc < 8; ++kc) {
      v16h f = ld_frag(&kb_[l16 * C_ + kc * 32], half * 8, 16 + half * 8);
      S0 = WMMA_F16(QA[kc], f, S0);
    }
#pragma unroll
    for (int kc = 0; kc < 8; ++kc) {
      v16h f = ld_frag(&kb_[(16 + l16) * C_ + kc * 32], half * 8, 16 + half * 8);
      S1 = WMMA_F16(QA[kc], f, S1);
    }

    // ---- online softmax over the 32 new keys (rows live on 16 lanes) ----
    float fr[8];
#pragma unroll
    for (int r = 0; r < 8; ++r) {
      float mx = fmaxf(S0[r], S1[r]);
      mx = fmaxf(mx, __shfl_xor(mx, 1, 32));
      mx = fmaxf(mx, __shfl_xor(mx, 2, 32));
      mx = fmaxf(mx, __shfl_xor(mx, 4, 32));
      mx = fmaxf(mx, __shfl_xor(mx, 8, 32));
      const float mnew = fmaxf(mrow[r], mx);
      fr[r] = __builtin_amdgcn_exp2f((mrow[r] - mnew) * C1f);
      const float p0 = __builtin_amdgcn_exp2f((S0[r] - mnew) * C1f);
      const float p1 = __builtin_amdgcn_exp2f((S1[r] - mnew) * C1f);
      S0[r] = p0; S1[r] = p1;
      float sm = p0 + p1;
      sm += __shfl_xor(sm, 1, 32);
      sm += __shfl_xor(sm, 2, 32);
      sm += __shfl_xor(sm, 4, 32);
      sm += __shfl_xor(sm, 8, 32);
      lrow[r] = lrow[r] * fr[r] + sm;
      mrow[r] = mnew;
    }
#pragma unroll
    for (int t = 0; t < 16; ++t)
#pragma unroll
      for (int r = 0; r < 8; ++r) O[t][r] *= fr[r];

    // ---- P: C/D layout -> A layout via per-wave LDS scratch ----
    _Float16* pw = pLds[wave];
#pragma unroll
    for (int r = 0; r < 8; ++r) {
      pw[(r + 8 * half) * 32 + l16]      = (_Float16)S0[r];
      pw[(r + 8 * half) * 32 + 16 + l16] = (_Float16)S1[r];
    }
    v16h PA = ld_frag(&pw[l16 * 32], half * 8, 16 + half * 8);

    // ---- O += P x V  (16 column tiles of 16 channels) ----
#pragma unroll
    for (int t = 0; t < 16; ++t) {
      v16h vf = ld_frag(&vb_[(t * 16 + l16) * 32], half * 8, 16 + half * 8);
      O[t] = WMMA_F16(PA, vf, O[t]);
    }
    __syncthreads();  // all waves done reading buf before it is refilled
    buf ^= 1;
  }

  // ---- normalize and write out[b][c][n] (f32) ----
#pragma unroll
  for (int t = 0; t < 16; ++t) {
    const int c = t * 16 + l16;
#pragma unroll
    for (int r = 0; r < 8; ++r) {
      const int qi = q0 + wave * 16 + r + 8 * half;
      out[((size_t)b * C_ + c) * N_ + qi] = O[t][r] / lrow[r];
    }
  }
}

extern "C" void kernel_launch(void* const* d_in, const int* in_sizes, int n_in,
                              void* d_out, int out_size, void* d_ws, size_t ws_size,
                              hipStream_t stream) {
  const float* x  = (const float*)d_in[0];
  const float* mo = (const float*)d_in[1];
  const float* wq = (const float*)d_in[2];
  const float* bq = (const float*)d_in[3];
  const float* wk = (const float*)d_in[4];
  const float* bk = (const float*)d_in[5];
  const float* wv = (const float*)d_in[6];
  const float* bv = (const float*)d_in[7];
  float* out = (float*)d_out;

  _Float16* qws = (_Float16*)d_ws;                    // [B][N][C] f16
  _Float16* kws = qws + (size_t)B_ * N_ * C_;         // [B][N][C] f16
  _Float16* vws = kws + (size_t)B_ * N_ * C_;         // [B][C][N] f16
  _Float16* wf  = vws + (size_t)B_ * N_ * C_;         // [3][C][C] f16

  wcvt_kernel<<<3 * C_ * C_ / 256, 256, 0, stream>>>(wq, wk, wv, wf);

  dim3 gp(N_ / 16, B_);
  qkv_proj<<<gp, 128, 0, stream>>>(x, mo, wf, bq, bk, bv, qws, kws, vws);

  dim3 ga(N_ / 128, B_);
  attn_kernel<<<ga, 256, 0, stream>>>(qws, kws, vws, out);
}
